// GRUGate_49941879718191
// MI455X (gfx1250) — matrix operand order repaired and checked by reading, not compiled, not run on hardware
//
#include <hip/hip_runtime.h>
#include <hip/hip_bf16.h>

typedef __attribute__((ext_vector_type(16))) __bf16 v16bf;
typedef __attribute__((ext_vector_type(8)))  __bf16 v8bf;
typedef __attribute__((ext_vector_type(8)))  float  v8f;

#define D_   1024
#define H_   1024
#define B_   16
#define L_   1024
#define KT_  32    // K-tiles per 1024 (K=32 per WMMA)
#define NT_  64    // N-tiles per 1024 (N=16 per WMMA)
#define TILE 512   // elements per 16x32 / 32x16 operand tile

#define WMMA_BF16(a, b, c) \
    __builtin_amdgcn_wmma_f32_16x16x32_bf16(false, (a), false, (b), (short)0, (c), false, false)

#if __has_builtin(__builtin_amdgcn_global_load_async_to_lds_b128)
#define HAVE_ASYNC_LDS 1
typedef int v4i_gcc __attribute__((__vector_size__(16)));
typedef __attribute__((address_space(1))) v4i_gcc as1_v4i;
typedef __attribute__((address_space(3))) v4i_gcc as3_v4i;
#endif

// Stage one A tile-row (KT_*TILE bf16 = 32 KB) into LDS with all 256 threads.
__device__ __forceinline__ void stage_a_lds(__bf16* lA, const __bf16* __restrict__ src) {
#if defined(HAVE_ASYNC_LDS)
#pragma unroll
    for (int i = 0; i < 8; ++i) {
        int off = ((int)threadIdx.x + i * 256) * 8;        // 16 bytes per op
        __builtin_amdgcn_global_load_async_to_lds_b128(
            (as1_v4i*)(uintptr_t)(src + off),
            (as3_v4i*)(uintptr_t)(lA + off), 0, 0);
    }
#if __has_builtin(__builtin_amdgcn_s_wait_asynccnt)
    __builtin_amdgcn_s_wait_asynccnt(0);
#else
    asm volatile("s_wait_asynccnt 0" ::: "memory");
#endif
#else
#pragma unroll
    for (int i = 0; i < 8; ++i) {
        int off = ((int)threadIdx.x + i * 256) * 8;
        *(v8bf*)(lA + off) = *(const v8bf*)(src + off);    // global_load + ds_store
    }
#endif
    __syncthreads();
}

// ---------------------------------------------------------------------------
// Swizzle helpers: pack fp32 sources into bf16 WMMA operand layout so that
// lane L of a wave loads its whole operand as elements [L*16, L*16+16).
// A (16x32, 16-bit): lane = khalf*16 + m ; elem j -> k = (j<8?0:16)+khalf*8+(j&7)
// B (32x16, 16-bit): lane = khalf*16 + n ; elem j -> k = (j<8?0:16)+khalf*8+(j&7)
// ---------------------------------------------------------------------------

// x (B,L,D) fp32 -> A tiles, row r = l*B + b  (B==16 so mtile == l, m == b)
__global__ void k_swizzleA_x(const float* __restrict__ x, __bf16* __restrict__ xA) {
    size_t idx = (size_t)blockIdx.x * blockDim.x + threadIdx.x;   // < 1024*32*512
    int within = (int)(idx & 511);
    size_t tile = idx >> 9;
    int lane = within >> 4, j = within & 15;
    int ktile = (int)(tile & (KT_ - 1));
    int mtile = (int)(tile >> 5);
    int m = lane & 15, khalf = lane >> 4;
    int kk = ((j < 8) ? 0 : 16) + khalf * 8 + (j & 7);
    int b = m, l = mtile;
    int d = ktile * 32 + kk;
    xA[idx] = (__bf16)x[((size_t)b * L_ + l) * D_ + d];
}

// h0 (B,H) fp32 -> single A tile-row (32 tiles)
__global__ void k_swizzleA_h0(const float* __restrict__ h0, __bf16* __restrict__ hA) {
    size_t idx = (size_t)blockIdx.x * blockDim.x + threadIdx.x;   // < 32*512
    int within = (int)(idx & 511);
    int ktile = (int)(idx >> 9);
    int lane = within >> 4, j = within & 15;
    int m = lane & 15, khalf = lane >> 4;
    int kk = ((j < 8) ? 0 : 16) + khalf * 8 + (j & 7);
    hA[idx] = (__bf16)h0[(size_t)m * H_ + ktile * 32 + kk];
}

// W rows [row0, row0+1024), 1024 cols fp32 -> B tiles, tile = ntile*KT + ktile
__global__ void k_swizzleB(const float* __restrict__ W, int row0, __bf16* __restrict__ dst) {
    size_t idx = (size_t)blockIdx.x * blockDim.x + threadIdx.x;   // < 64*32*512
    int within = (int)(idx & 511);
    size_t tile = idx >> 9;
    int ktile = (int)(tile & (KT_ - 1));
    int ntile = (int)(tile >> 5);
    int lane = within >> 4, j = within & 15;
    int n = lane & 15;
    int k = ((j < 8) ? 0 : 16) + (lane >> 4) * 8 + (j & 7);
    dst[idx] = (__bf16)W[(size_t)(row0 + ktile * 32 + k) * 1024 + ntile * 16 + n];
}

// ---------------------------------------------------------------------------
// Phase 1: xg[g][l*B+b][h] = x @ W{r,z,n}_x + bias   (bf16 WMMA, fp32 acc)
// block = (mtile, gate, quarter): A row staged in LDS, shared by 8 waves;
// wave = 2 N-tiles; explicit even/odd double-buffer (acc16+A16+B32 VGPRs,
// fits without spills while keeping stage k+1 loads in flight over stage k)
// ---------------------------------------------------------------------------
__global__ void k_gemm_pre(const __bf16* __restrict__ xA, const __bf16* __restrict__ WxB,
                           const float* __restrict__ br, const float* __restrict__ bz,
                           const float* __restrict__ bn, float* __restrict__ xg) {
    __shared__ __bf16 lA[KT_ * TILE];
    int wid = threadIdx.x >> 5, lane = threadIdx.x & 31;
    int quarter = blockIdx.x & 3;
    int g       = (blockIdx.x >> 2) % 3;
    int mtile   = blockIdx.x / 12;
    int ngroup  = quarter * 8 + wid;          // 0..31, pair of N-tiles

    stage_a_lds(lA, xA + (size_t)mtile * KT_ * TILE);

    const __bf16* aL    = lA + lane * 16;
    const __bf16* bBase = WxB + (((size_t)g * NT_ + ngroup * 2) * KT_) * TILE + lane * 16;

    v8f acc[2] = {v8f{}, v8f{}};
    v16bf a0 = *(const v16bf*)(aL);
    v16bf a1 = *(const v16bf*)(aL + TILE);
    v16bf b0[2], b1[2];
#pragma unroll
    for (int i = 0; i < 2; ++i) {
        b0[i] = *(const v16bf*)(bBase + ((size_t)i * KT_ + 0) * TILE);
        b1[i] = *(const v16bf*)(bBase + ((size_t)i * KT_ + 1) * TILE);
    }

#pragma clang loop unroll(disable)
    for (int kt = 0; kt < KT_ - 2; kt += 2) {
#pragma unroll
        for (int i = 0; i < 2; ++i) acc[i] = WMMA_BF16(a0, b0[i], acc[i]);
        a0 = *(const v16bf*)(aL + (size_t)(kt + 2) * TILE);
#pragma unroll
        for (int i = 0; i < 2; ++i)
            b0[i] = *(const v16bf*)(bBase + ((size_t)i * KT_ + kt + 2) * TILE);
        __builtin_prefetch(bBase + (size_t)(kt + 4) * TILE, 0, 3);
#pragma unroll
        for (int i = 0; i < 2; ++i) acc[i] = WMMA_BF16(a1, b1[i], acc[i]);
        a1 = *(const v16bf*)(aL + (size_t)(kt + 3) * TILE);
#pragma unroll
        for (int i = 0; i < 2; ++i)
            b1[i] = *(const v16bf*)(bBase + ((size_t)i * KT_ + kt + 3) * TILE);
    }
#pragma unroll
    for (int i = 0; i < 2; ++i) acc[i] = WMMA_BF16(a0, b0[i], acc[i]);
#pragma unroll
    for (int i = 0; i < 2; ++i) acc[i] = WMMA_BF16(a1, b1[i], acc[i]);

    const float* bias = (g == 0) ? br : ((g == 1) ? bz : bn);
    int n = lane & 15, m0 = (lane >> 4) * 8;
    float* y = xg + (size_t)g * ((size_t)L_ * B_ * H_);
#pragma unroll
    for (int i = 0; i < 2; ++i) {
        int col = (ngroup * 2 + i) * 16 + n;
        float bv = bias[col];
#pragma unroll
        for (int v = 0; v < 8; ++v)
            y[(size_t)(mtile * 16 + m0 + v) * H_ + col] = acc[i][v] + bv;
    }
}

__device__ __forceinline__ float sigmoid_f(float v) { return 1.0f / (1.0f + __expf(-v)); }
__device__ __forceinline__ float tanh_f(float v) {
    v = fminf(fmaxf(v, -15.0f), 15.0f);
    float e = __expf(2.0f * v);
    return (e - 1.0f) / (e + 1.0f);
}

// ---------------------------------------------------------------------------
// Phase 2: one GRU step. 8 blocks x 8 waves; block stages h_{t-1} A-row in
// LDS; wave = one 16-col tile of H across all 3 gates; gate fusion in-wave.
// Writes h_t as fp32 (hs) AND pre-swizzled bf16 A-tiles (hsA).
// ---------------------------------------------------------------------------
__global__ void k_gru_step(const __bf16* __restrict__ aPrev, const __bf16* __restrict__ WhB,
                           const float* __restrict__ hPrev,
                           const float* __restrict__ xr_t, const float* __restrict__ xz_t,
                           const float* __restrict__ xn_t,
                           float* __restrict__ hs_t, __bf16* __restrict__ hsA_t) {
    __shared__ __bf16 lA[KT_ * TILE];
    int wid = threadIdx.x >> 5, lane = threadIdx.x & 31;
    int ntile = blockIdx.x * 8 + wid;         // 0..63

    stage_a_lds(lA, aPrev);

    const __bf16* aL = lA + lane * 16;
    const __bf16* bB[3];
#pragma unroll
    for (int gi = 0; gi < 3; ++gi)
        bB[gi] = WhB + (((size_t)gi * NT_ + ntile) * KT_) * TILE + lane * 16;

    v8f acc[3] = {v8f{}, v8f{}, v8f{}};
    v16bf a0 = *(const v16bf*)(aL);
    v16bf a1 = *(const v16bf*)(aL + TILE);
    v16bf b0[3], b1[3];
#pragma unroll
    for (int gi = 0; gi < 3; ++gi) {
        b0[gi] = *(const v16bf*)(bB[gi]);
        b1[gi] = *(const v16bf*)(bB[gi] + TILE);
    }

#pragma clang loop unroll(disable)
    for (int kt = 0; kt < KT_ - 2; kt += 2) {
#pragma unroll
        for (int gi = 0; gi < 3; ++gi) acc[gi] = WMMA_BF16(a0, b0[gi], acc[gi]);
        a0 = *(const v16bf*)(aL + (size_t)(kt + 2) * TILE);
#pragma unroll
        for (int gi = 0; gi < 3; ++gi)
            b0[gi] = *(const v16bf*)(bB[gi] + (size_t)(kt + 2) * TILE);
#pragma unroll
        for (int gi = 0; gi < 3; ++gi) acc[gi] = WMMA_BF16(a1, b1[gi], acc[gi]);
        a1 = *(const v16bf*)(aL + (size_t)(kt + 3) * TILE);
#pragma unroll
        for (int gi = 0; gi < 3; ++gi)
            b1[gi] = *(const v16bf*)(bB[gi] + (size_t)(kt + 3) * TILE);
    }
#pragma unroll
    for (int gi = 0; gi < 3; ++gi) acc[gi] = WMMA_BF16(a0, b0[gi], acc[gi]);
#pragma unroll
    for (int gi = 0; gi < 3; ++gi) acc[gi] = WMMA_BF16(a1, b1[gi], acc[gi]);

    int n = lane & 15, m0 = (lane >> 4) * 8;
    int col = ntile * 16 + n;
#pragma unroll
    for (int v = 0; v < 8; ++v) {
        int m = m0 + v;
        size_t idx = (size_t)m * H_ + col;
        float rg = sigmoid_f(xr_t[idx] + acc[0][v]);
        float zg = sigmoid_f(xz_t[idx] + acc[1][v]);
        float ng = tanh_f(xn_t[idx] + rg * acc[2][v]);
        float hn = (1.0f - zg) * hPrev[idx] + zg * ng;
        hs_t[idx] = hn;
        // scatter into swizzled A layout for next step / output GEMM
        int kt2 = col >> 5, kk = col & 31;
        int hi = kk >> 4, rem = kk & 15;
        int khalf = rem >> 3, j = hi * 8 + (rem & 7);
        hsA_t[((size_t)kt2) * TILE + (khalf * 16 + m) * 16 + j] = (__bf16)hn;
    }
}

// ---------------------------------------------------------------------------
// Phase 3: out[b][l][d] = hs[l][b][:] @ Wo + bo    (mtile == l, m == b)
// block = (mtile, quarter): A row staged in LDS, shared by 8 waves
// ---------------------------------------------------------------------------
__global__ void k_gemm_out(const __bf16* __restrict__ hsA, const __bf16* __restrict__ WoB,
                           const float* __restrict__ bo, float* __restrict__ out) {
    __shared__ __bf16 lA[KT_ * TILE];
    int wid = threadIdx.x >> 5, lane = threadIdx.x & 31;
    int quarter = blockIdx.x & 3;
    int mtile   = blockIdx.x >> 2;
    int ngroup  = quarter * 8 + wid;          // 0..31, pair of N-tiles

    stage_a_lds(lA, hsA + (size_t)mtile * KT_ * TILE);

    const __bf16* aL    = lA + lane * 16;
    const __bf16* bBase = WoB + (((size_t)(ngroup * 2)) * KT_) * TILE + lane * 16;

    v8f acc[2] = {v8f{}, v8f{}};
    v16bf a0 = *(const v16bf*)(aL);
    v16bf a1 = *(const v16bf*)(aL + TILE);
    v16bf b0[2], b1[2];
#pragma unroll
    for (int i = 0; i < 2; ++i) {
        b0[i] = *(const v16bf*)(bBase + ((size_t)i * KT_ + 0) * TILE);
        b1[i] = *(const v16bf*)(bBase + ((size_t)i * KT_ + 1) * TILE);
    }

#pragma clang loop unroll(disable)
    for (int kt = 0; kt < KT_ - 2; kt += 2) {
#pragma unroll
        for (int i = 0; i < 2; ++i) acc[i] = WMMA_BF16(a0, b0[i], acc[i]);
        a0 = *(const v16bf*)(aL + (size_t)(kt + 2) * TILE);
#pragma unroll
        for (int i = 0; i < 2; ++i)
            b0[i] = *(const v16bf*)(bBase + ((size_t)i * KT_ + kt + 2) * TILE);
        __builtin_prefetch(bBase + (size_t)(kt + 4) * TILE, 0, 3);
#pragma unroll
        for (int i = 0; i < 2; ++i) acc[i] = WMMA_BF16(a1, b1[i], acc[i]);
        a1 = *(const v16bf*)(aL + (size_t)(kt + 3) * TILE);
#pragma unroll
        for (int i = 0; i < 2; ++i)
            b1[i] = *(const v16bf*)(bBase + ((size_t)i * KT_ + kt + 3) * TILE);
    }
#pragma unroll
    for (int i = 0; i < 2; ++i) acc[i] = WMMA_BF16(a0, b0[i], acc[i]);
#pragma unroll
    for (int i = 0; i < 2; ++i) acc[i] = WMMA_BF16(a1, b1[i], acc[i]);

    int n = lane & 15, m0 = (lane >> 4) * 8;
#pragma unroll
    for (int i = 0; i < 2; ++i) {
        int d = (ngroup * 2 + i) * 16 + n;
        float bv = bo[d];
#pragma unroll
        for (int v = 0; v < 8; ++v) {
            int b = m0 + v;              // batch
            int l = mtile;               // time step
            out[((size_t)b * L_ + l) * D_ + d] = acc[i][v] + bv;
        }
    }
}

// ---------------------------------------------------------------------------
extern "C" void kernel_launch(void* const* d_in, const int* in_sizes, int n_in,
                              void* d_out, int out_size, void* d_ws, size_t ws_size,
                              hipStream_t stream) {
    const float* x      = (const float*)d_in[0];
    const float* h_prev = (const float*)d_in[1];
    const float* Wr     = (const float*)d_in[2];
    const float* br     = (const float*)d_in[3];
    const float* Wz     = (const float*)d_in[4];
    const float* bz     = (const float*)d_in[5];
    const float* Wn     = (const float*)d_in[6];
    const float* bn     = (const float*)d_in[7];
    const float* Wo     = (const float*)d_in[8];
    const float* bo     = (const float*)d_in[9];
    float* out = (float*)d_out;

    const size_t EL_xA  = (size_t)1024 * KT_ * TILE;       // 16,777,216 bf16
    const size_t EL_BM  = (size_t)NT_ * KT_ * TILE;        // 1,048,576 bf16 per matrix
    const size_t EL_xg  = (size_t)3 * L_ * B_ * H_;        // fp32
    const size_t EL_hs  = (size_t)L_ * B_ * H_;            // fp32
    const size_t EL_hsA = (size_t)1024 * KT_ * TILE;       // bf16
    const size_t EL_h0A = (size_t)KT_ * TILE;              // bf16

    char* p = (char*)d_ws;
    __bf16* xA  = (__bf16*)p;  p += EL_xA  * sizeof(__bf16);
    __bf16* WxB = (__bf16*)p;  p += 3 * EL_BM * sizeof(__bf16);
    __bf16* WhB = (__bf16*)p;  p += 3 * EL_BM * sizeof(__bf16);
    __bf16* WoB = (__bf16*)p;  p += EL_BM  * sizeof(__bf16);
    __bf16* hsA = (__bf16*)p;  p += EL_hsA * sizeof(__bf16);
    __bf16* h0A = (__bf16*)p;  p += EL_h0A * sizeof(__bf16);
    float*  xg  = (float*)p;   p += EL_xg  * sizeof(float);
    float*  hs  = (float*)p;   p += EL_hs  * sizeof(float);

    // --- operand packing ---
    k_swizzleA_x <<<65536, 256, 0, stream>>>(x, xA);
    k_swizzleA_h0<<<64,    256, 0, stream>>>(h_prev, h0A);
    k_swizzleB   <<<4096,  256, 0, stream>>>(Wr,    0, WxB + 0 * EL_BM);
    k_swizzleB   <<<4096,  256, 0, stream>>>(Wz,    0, WxB + 1 * EL_BM);
    k_swizzleB   <<<4096,  256, 0, stream>>>(Wn,    0, WxB + 2 * EL_BM);
    k_swizzleB   <<<4096,  256, 0, stream>>>(Wr, 1024, WhB + 0 * EL_BM);
    k_swizzleB   <<<4096,  256, 0, stream>>>(Wz, 1024, WhB + 1 * EL_BM);
    k_swizzleB   <<<4096,  256, 0, stream>>>(Wn, 1024, WhB + 2 * EL_BM);
    k_swizzleB   <<<4096,  256, 0, stream>>>(Wo,    0, WoB);

    // --- phase 1: input projections ---
    k_gemm_pre<<<12288, 256, 0, stream>>>(xA, WxB, br, bz, bn, xg);

    // --- phase 2: sequential scan ---
    const size_t S  = (size_t)L_ * B_ * H_;
    const size_t BH = (size_t)B_ * H_;
    for (int t = 0; t < L_; ++t) {
        const __bf16* aPrev = (t == 0) ? h0A : (hsA + (size_t)(t - 1) * KT_ * TILE);
        const float*  hp    = (t == 0) ? h_prev : (hs + (size_t)(t - 1) * BH);
        k_gru_step<<<8, 256, 0, stream>>>(aPrev, WhB, hp,
                                          xg + 0 * S + (size_t)t * BH,
                                          xg + 1 * S + (size_t)t * BH,
                                          xg + 2 * S + (size_t)t * BH,
                                          hs + (size_t)t * BH,
                                          hsA + (size_t)t * KT_ * TILE);
    }

    // --- phase 3: output projection ---
    k_gemm_out<<<4096, 256, 0, stream>>>(hsA, WoB, bo, out);

    // h_final appended after outputs
    (void)hipMemcpyAsync(out + (size_t)B_ * L_ * D_, hs + (size_t)(L_ - 1) * BH,
                         BH * sizeof(float), hipMemcpyDeviceToDevice, stream);
}
